// M_34514357191211
// MI455X (gfx1250) — matrix operand order repaired and checked by reading, not compiled, not run on hardware
//
#include <hip/hip_runtime.h>
#include <hip/hip_bf16.h>
#include <stdint.h>

// ---------------------------------------------------------------------------
// Fused residual-add + LayerNorm(C=384) + Linear(384->1536, bf16 WMMA) +
// bias + exact GELU for MI455X (gfx1250, wave32, WMMA 16x16x32 bf16).
//
// Roofline: 118 GFLOP GEMM vs ~930MB minimal HBM traffic (~40us @ 23.3TB/s)
// -> fuse everything; x read once, normalized A panel held in LDS as bf16,
// W (2.4MB) converted once to bf16 in d_ws and streamed from L2.
// Epilogue stores go through a wave-private LDS transpose so global writes
// are 128B-contiguous global_store_b128 instead of strided b32 stores.
// ---------------------------------------------------------------------------

typedef __attribute__((ext_vector_type(16))) __bf16 v16bf;
typedef __attribute__((ext_vector_type(8)))  float  v8f;

#define ROWS_PER_BLK 64          // M tile (rows of x)
#define F_TILE       128         // N tile (output features per inner pass)
#define C_DIM        384         // K
#define F_DIM        1536
#define LDS_ROW_DW   196         // 384 bf16 = 192 dwords, +4 dwords pad
#define A_DWORDS     (ROWS_PER_BLK * LDS_ROW_DW)       // 12544 dwords
#define STAGE_STRIDE 36          // 32 floats + 4 pad (16B-aligned rows)
#define STAGE_FLOATS (8 * 32 * STAGE_STRIDE)           // 9216 floats
#define SMEM_BYTES   (A_DWORDS * 4 + STAGE_FLOATS * 4) // 87040 bytes

union FragU {
    uint4  u[2];                 // two 16-byte halves (8 VGPRs)
    v16bf  v;
};

__device__ __forceinline__ uint32_t pack_bf16x2(float lo, float hi) {
    uint32_t a = __float_as_uint(lo);
    uint32_t b = __float_as_uint(hi);
    uint32_t al = (a + 0x7FFFu + ((a >> 16) & 1u)) >> 16;   // RNE
    uint32_t bh = (b + 0x7FFFu + ((b >> 16) & 1u)) >> 16;
    return al | (bh << 16);
}

// ---------------------------------------------------------------------------
// Kernel 1: convert Wl (F_DIM x C_DIM fp32, row-major) -> bf16 in workspace.
// ---------------------------------------------------------------------------
__global__ __launch_bounds__(256) void convert_w_bf16(const float* __restrict__ Wl,
                                                      uint32_t* __restrict__ Wb) {
    int i = blockIdx.x * blockDim.x + threadIdx.x;           // float2 index
    const float2* src = (const float2*)Wl;
    float2 v = src[i];
    Wb[i] = pack_bf16x2(v.x, v.y);
}

// ---------------------------------------------------------------------------
// Kernel 2: fused add + LN + GEMM + bias + GELU.
// Grid: 1568 blocks (100352 rows / 64). Block: 256 threads = 8 waves.
// ---------------------------------------------------------------------------
__global__ __launch_bounds__(256) void fused_ln_gemm_gelu(
        const float* __restrict__ x355,
        const float* __restrict__ x369,
        const float* __restrict__ gamma,
        const float* __restrict__ beta,
        const uint32_t* __restrict__ Wb,   // bf16 pairs, [F_DIM][192] dwords
        const float* __restrict__ bl,
        float* __restrict__ out) {

    extern __shared__ char smemRaw[];
    uint32_t* shA     = (uint32_t*)smemRaw;                  // 64 x 196 dwords
    float*    shStage = (float*)(smemRaw + A_DWORDS * 4);    // 8 waves x 32x36

    const int tid    = threadIdx.x;
    const int lane   = tid & 31;
    const int wid    = tid >> 5;          // 0..7
    const int laneLo = lane & 15;
    const int sel    = lane >> 4;         // 0: lanes 0-15, 1: lanes 16-31
    const int rowBase = blockIdx.x * ROWS_PER_BLK;

    // ---------------- Phase 1: residual add + LayerNorm -> bf16 LDS panel ----
    const float4* g4 = (const float4*)gamma;
    const float4* b4 = (const float4*)beta;
    float4 gv[3], bv[3];
    #pragma unroll
    for (int c = 0; c < 3; ++c) {
        gv[c] = g4[lane + 32 * c];
        bv[c] = b4[lane + 32 * c];
    }

    for (int j = 0; j < 8; ++j) {
        const int r = wid * 8 + j;                       // local row 0..63
        const size_t gr = (size_t)(rowBase + r) * C_DIM;
        const float4* p0 = (const float4*)(x355 + gr);
        const float4* p1 = (const float4*)(x369 + gr);

        float4 v[3];
        float s = 0.f, s2 = 0.f;
        #pragma unroll
        for (int c = 0; c < 3; ++c) {
            float4 a = p0[lane + 32 * c];
            float4 b = p1[lane + 32 * c];
            v[c].x = a.x + b.x; v[c].y = a.y + b.y;
            v[c].z = a.z + b.z; v[c].w = a.w + b.w;
            s  += v[c].x + v[c].y + v[c].z + v[c].w;
            s2 += v[c].x * v[c].x + v[c].y * v[c].y +
                  v[c].z * v[c].z + v[c].w * v[c].w;
        }
        #pragma unroll
        for (int off = 16; off > 0; off >>= 1) {
            s  += __shfl_xor(s,  off);
            s2 += __shfl_xor(s2, off);
        }
        const float mu   = s * (1.0f / C_DIM);
        const float var  = s2 * (1.0f / C_DIM) - mu * mu;
        const float rstd = __frsqrt_rn(var + 1e-5f);

        uint32_t* Ar = &shA[r * LDS_ROW_DW];
        #pragma unroll
        for (int c = 0; c < 3; ++c) {
            float n0 = (v[c].x - mu) * rstd * gv[c].x + bv[c].x;
            float n1 = (v[c].y - mu) * rstd * gv[c].y + bv[c].y;
            float n2 = (v[c].z - mu) * rstd * gv[c].z + bv[c].z;
            float n3 = (v[c].w - mu) * rstd * gv[c].w + bv[c].w;
            uint2 pk;
            pk.x = pack_bf16x2(n0, n1);
            pk.y = pack_bf16x2(n2, n3);
            *(uint2*)(Ar + (lane + 32 * c) * 2) = pk;    // k-dwords 2i, 2i+1
        }
    }
    __syncthreads();

    // ---------------- Phase 2: bf16 WMMA GEMM + bias + GELU ------------------
    // Wave grid 2(M) x 4(N); each wave: 2x2 subtiles of 16x16 (32x32 tile).
    const int waveM = wid >> 2;           // 0..1 -> 32-row half
    const int waveN = wid & 3;            // 0..3 -> 32-col quarter
    float* st = shStage + wid * 32 * STAGE_STRIDE;       // wave-private stage

    const int c4   = lane & 7;            // readback: float4 column 0..7
    const int rsub = lane >> 3;           // readback: row sub-index 0..3

    for (int ft = 0; ft < F_DIM / F_TILE; ++ft) {
        const int fBase = ft * F_TILE;

        v8f acc[2][2];
        #pragma unroll
        for (int mi = 0; mi < 2; ++mi)
            #pragma unroll
            for (int ni = 0; ni < 2; ++ni)
                acc[mi][ni] = {};

        #pragma unroll 4
        for (int kk = 0; kk < C_DIM / 32; ++kk) {
            const int kd = kk * 16;       // dword offset of this 32-wide k step

            // A fragments from LDS (16x32 bf16, ISA lane striping):
            // lanes 0-15: row M=lane, VGPR0-3 = K0-7, VGPR4-7 = K16-23
            // lanes16-31: row M=lane-16, VGPR0-3 = K8-15, VGPR4-7 = K24-31
            FragU afr[2];
            #pragma unroll
            for (int mi = 0; mi < 2; ++mi) {
                const int r = (waveM * 2 + mi) * 16 + laneLo;
                const uint32_t* p = &shA[r * LDS_ROW_DW + kd + sel * 4];
                afr[mi].u[0] = *(const uint4*)(p);        // ds_load_b128
                afr[mi].u[1] = *(const uint4*)(p + 8);    // ds_load_b128
            }

            // B fragments straight from global bf16 W (rows of Wl are the
            // k-contiguous columns of B): lanes 0-15 = K0-7,16-23; 16-31 = K8-15,24-31.
            FragU bfr[2];
            #pragma unroll
            for (int ni = 0; ni < 2; ++ni) {
                const int f = fBase + (waveN * 2 + ni) * 16 + laneLo;
                const uint32_t* p = Wb + (size_t)f * (C_DIM / 2) + kd + sel * 4;
                bfr[ni].u[0] = *(const uint4*)(p);        // global_load_b128
                bfr[ni].u[1] = *(const uint4*)(p + 8);    // global_load_b128
            }

            #pragma unroll
            for (int mi = 0; mi < 2; ++mi)
                #pragma unroll
                for (int ni = 0; ni < 2; ++ni)
                    acc[mi][ni] = __builtin_amdgcn_wmma_f32_16x16x32_bf16(
                        false, afr[mi].v, false, bfr[ni].v,
                        (short)0, acc[mi][ni], false, false);
        }

        // Epilogue: bias + exact GELU in accumulator layout, stage the wave's
        // 32x32 tile into private LDS, then store 128B-contiguous b128 rows.
        // C/D layout: lanes 0-15 -> N=lane, VGPR r -> M=r; lanes 16-31 -> M=8+r.
        float bias[2];
        #pragma unroll
        for (int ni = 0; ni < 2; ++ni)
            bias[ni] = bl[fBase + (waveN * 2 + ni) * 16 + laneLo];

        #pragma unroll
        for (int mi = 0; mi < 2; ++mi) {
            #pragma unroll
            for (int ni = 0; ni < 2; ++ni) {
                const int col  = ni * 16 + laneLo;
                const int row0 = mi * 16 + sel * 8;
                #pragma unroll
                for (int r = 0; r < 8; ++r) {
                    float y = acc[mi][ni][r] + bias[ni];
                    float g = 0.5f * y * (1.0f + erff(y * 0.70710678118654752f));
                    st[(row0 + r) * STAGE_STRIDE + col] = g;   // ds_store_b32
                }
            }
        }
        __builtin_amdgcn_wave_barrier();   // LDS in-order within wave; pin sched

        // Readback: store-instr i writes rows {4i..4i+3}, each as a 128B
        // contiguous segment (lanes grouped 8-per-row, float4 each).
        const int gcol = fBase + waveN * 32 + c4 * 4;
        #pragma unroll
        for (int i = 0; i < 8; ++i) {
            const int row = i * 4 + rsub;
            float4 vv = *(const float4*)&st[row * STAGE_STRIDE + c4 * 4];
            const size_t grow = (size_t)(rowBase + waveM * 32 + row);
            *(float4*)(out + grow * F_DIM + gcol) = vv;        // global_store_b128
        }
        __builtin_amdgcn_wave_barrier();   // don't overwrite stage early
    }
}

// ---------------------------------------------------------------------------
extern "C" void kernel_launch(void* const* d_in, const int* in_sizes, int n_in,
                              void* d_out, int out_size, void* d_ws, size_t ws_size,
                              hipStream_t stream) {
    (void)in_sizes; (void)n_in; (void)out_size; (void)ws_size;
    const float* x355  = (const float*)d_in[0];
    const float* x369  = (const float*)d_in[1];
    const float* gamma = (const float*)d_in[2];
    const float* beta  = (const float*)d_in[3];
    const float* Wl    = (const float*)d_in[4];
    const float* bl    = (const float*)d_in[5];
    float*       out   = (float*)d_out;
    uint32_t*    Wb    = (uint32_t*)d_ws;   // 1536*384 bf16 = 1.18 MB

    // 1536*384/2 = 294912 float2 -> dword conversions
    convert_w_bf16<<<294912 / 256, 256, 0, stream>>>(Wl, Wb);

    // 100352 rows / 64 per block = 1568 blocks; 87KB dynamic LDS
    fused_ln_gemm_gelu<<<1568, 256, SMEM_BYTES, stream>>>(x355, x369, gamma, beta,
                                                          Wb, bl, out);
}
